// SelfTransformerBlock_84628035601169
// MI455X (gfx1250) — compile-verified, matrix operand order checked by reading, not run on hardware
//
#include <hip/hip_runtime.h>
#include <math.h>

// ---------------- CDNA5 types ----------------
typedef __bf16 bf16_t;
typedef __attribute__((ext_vector_type(16))) __bf16 v16bf;
typedef __attribute__((ext_vector_type(8)))  float  v8f;
typedef __attribute__((ext_vector_type(4)))  float  f32x4;

union FragU { v16bf v; f32x4 f[2]; };

static __device__ __forceinline__ v16bf load_frag(const bf16_t* p) {
    FragU r;
    r.f[0] = *(const f32x4*)(p);
    r.f[1] = *(const f32x4*)(p + 16);
    return r.v;
}

static __device__ __forceinline__ float eluf(float x) {
    return x > 0.0f ? x : (expf(x) - 1.0f);
}

// ---------------- async global->LDS (CDNA5 ASYNCcnt path) ----------------
#if defined(__has_builtin)
#  if __has_builtin(__builtin_amdgcn_global_load_async_to_lds_b128)
#    define USE_ASYNC_LDS 1
#  endif
#endif
#ifndef USE_ASYNC_LDS
#  define USE_ASYNC_LDS 0
#endif

// builtin expects: (int4 addrspace(1)* src, int4 addrspace(3)* dst, imm offset, imm cpol)
typedef int gvec_i32x4 __attribute__((vector_size(16)));
typedef __attribute__((address_space(1))) gvec_i32x4 as1_i32x4;
typedef __attribute__((address_space(3))) gvec_i32x4 as3_i32x4;

static __device__ __forceinline__ void async_copy16(const bf16_t* g, bf16_t* l) {
#if USE_ASYNC_LDS
    __builtin_amdgcn_global_load_async_to_lds_b128((as1_i32x4*)g, (as3_i32x4*)l, 0, 0);
#else
    *(f32x4*)l = *(const f32x4*)g;
#endif
}

#if USE_ASYNC_LDS
#  if __has_builtin(__builtin_amdgcn_s_wait_asynccnt)
#    define WAIT_ASYNC(n) __builtin_amdgcn_s_wait_asynccnt(n)
#  else
#    define WAIT_ASYNC(n) asm volatile("s_wait_asynccnt %0" :: "i"(n) : "memory")
#  endif
#else
#  define WAIT_ASYNC(n) ((void)0)
#endif

// Problem constants
#define BB   4
#define NN   4096
#define CC   512
#define HH   8
#define DH   64
#define MM   (BB * NN)          // 16384 rows
#define KCH  64                 // K chunk staged in LDS per buffer
#define BST  72                 // LDS row pitch (bank-conflict padding)

// ---------------- generic GEMM: C = A(bf16,[M,K]) * W(bf16,[N,K])^T ----------------
// block = 256 thr = 8 waves; wave -> 16 rows x 64 cols; grid = (M/128, N/64).
// Weight tile (64 x KCH) shared by all 8 waves is staged in LDS via async
// global->LDS DMA, double buffered against the WMMA stream.
template <int RELU>
__global__ __launch_bounds__(256) void gemm_bf16_nt(
    const bf16_t* __restrict__ A, int lda,
    const bf16_t* __restrict__ W, int ldb,
    bf16_t* __restrict__ C, int ldc, int K)
{
    __shared__ __align__(16) bf16_t Bsm[2][64 * BST];

    const int wave = threadIdx.x >> 5;
    const int lane = threadIdx.x & 31;
    const int m0 = blockIdx.x * 128 + wave * 16;
    const int n0 = blockIdx.y * 64;
    const int row = lane & 15;
    const int kb  = (lane >> 4) * 8;

    v8f acc[4] = {};
    const bf16_t* ap = A + (size_t)(m0 + row) * lda + kb;

    // one stage = 64 rows x KCH cols = 512 16B chunks; 256 threads x 2 chunks
    auto issue_stage = [&](int buf, int k0) {
#pragma unroll
        for (int ii = 0; ii < 2; ++ii) {
            int i  = threadIdx.x + ii * 256;
            int r  = i >> 3;
            int ko = (i & 7) * 8;
            async_copy16(W + (size_t)(n0 + r) * ldb + k0 + ko,
                         &Bsm[buf][r * BST + ko]);
        }
    };

    const int nch = K / KCH;
    issue_stage(0, 0);
    for (int kc = 0; kc < nch; ++kc) {
        const int buf = kc & 1;
        if (kc + 1 < nch) {
            issue_stage(buf ^ 1, (kc + 1) * KCH);
            WAIT_ASYNC(2);      // 2 outstanding = next stage only -> current done
        } else {
            WAIT_ASYNC(0);
        }
        __syncthreads();        // current stage visible to all waves

        const bf16_t* bs = &Bsm[buf][0];
        const int k0 = kc * KCH;
#pragma unroll
        for (int kk = 0; kk < KCH; kk += 32) {
            v16bf a = load_frag(ap + k0 + kk);
#pragma unroll
            for (int j = 0; j < 4; ++j) {
                v16bf bfr = load_frag(bs + (j * 16 + row) * BST + kb + kk);
                acc[j] = __builtin_amdgcn_wmma_f32_16x16x32_bf16(false, a, false, bfr, (short)0, acc[j], false, false);
            }
        }
        __syncthreads();        // done reading buf before it is overwritten
    }

    const int crow = (lane >> 4) * 8;   // C/D layout: lanes16-31 hold M=8..15
    const int ccol = lane & 15;
#pragma unroll
    for (int j = 0; j < 4; ++j) {
#pragma unroll
        for (int r = 0; r < 8; ++r) {
            float v = acc[j][r];
            if (RELU) v = v > 0.0f ? v : 0.0f;
            C[(size_t)(m0 + crow + r) * ldc + (n0 + j * 16 + ccol)] = (bf16_t)v;
        }
    }
}

// ---------------- weight f32 -> bf16 ----------------
__global__ void cvt_f32_bf16(const float* __restrict__ s, bf16_t* __restrict__ d, int n)
{
    int i = blockIdx.x * blockDim.x + threadIdx.x;
    if (i < n) d[i] = (bf16_t)s[i];
}

// ---------------- x [B,C,N] -> ffin[:, 0:512] bf16 [B*N, 1024] ----------------
__global__ void x_to_ffin(const float* __restrict__ x, bf16_t* __restrict__ ffin)
{
    int idx = blockIdx.x * blockDim.x + threadIdx.x;   // B*N*C threads
    int c = idx & (CC - 1);
    int n = (idx >> 9) & (NN - 1);
    int b = idx >> 21;
    float v = x[((size_t)(b * CC + c)) * NN + n];
    ffin[((size_t)(b * NN + n)) * (2 * CC) + c] = (bf16_t)v;
}

// ---------------- feature map: elu+1, mask; build Qp, K^T, V^T ----------------
__global__ void feature_map(const bf16_t* __restrict__ QKV, const float* __restrict__ mask,
                            bf16_t* __restrict__ Qp, bf16_t* __restrict__ KpT,
                            bf16_t* __restrict__ Vt)
{
    int idx = blockIdx.x * blockDim.x + threadIdx.x;   // B*N*C
    int col = idx & (CC - 1);
    int n   = (idx >> 9) & (NN - 1);
    int b   = idx >> 21;
    int h = col >> 6, d = col & 63;
    size_t rowq = (size_t)(b * NN + n) * (3 * CC);
    float q = (float)QKV[rowq + col];
    float k = (float)QKV[rowq + CC + col];
    float v = (float)QKV[rowq + 2 * CC + col];
    float mk = mask[b * NN + n];
    float qp = (eluf(q) + 1.0f) * mk;
    float kp = (eluf(k) + 1.0f) * mk;
    v *= mk;
    Qp[(size_t)(b * NN + n) * CC + col] = (bf16_t)qp;
    size_t tIdx = ((size_t)(b * HH + h) * DH + d) * NN + n;
    KpT[tIdx] = (bf16_t)kp;
    Vt[tIdx]  = (bf16_t)v;
}

// ---------------- Ksum[b,h,d] = sum_n Kp ----------------
__global__ void ksum_kernel(const bf16_t* __restrict__ KpT, float* __restrict__ Ksum)
{
    int t = blockIdx.x * blockDim.x + threadIdx.x;     // 32*64 threads
    const bf16_t* p = KpT + (size_t)t * NN;
    float s = 0.0f;
    for (int i = 0; i < NN; ++i) s += (float)p[i];
    Ksum[t] = s;
}

// ---------------- Z[b,n,h] = 1/(Q . Ksum + eps) ----------------
__global__ void z_kernel(const bf16_t* __restrict__ Qp, const float* __restrict__ Ksum,
                         float* __restrict__ Z)
{
    int t = blockIdx.x * blockDim.x + threadIdx.x;     // B*N*H
    int h = t & 7;
    size_t rg = (size_t)(t >> 3);                      // b*N + n
    int b = (int)(rg >> 12);
    const bf16_t* q = Qp + rg * CC + h * DH;
    const float* ks = Ksum + (b * HH + h) * DH;
    float s = 0.0f;
#pragma unroll 8
    for (int d = 0; d < DH; ++d) s += (float)q[d] * ks[d];
    Z[t] = 1.0f / (s + 1e-6f);
}

// ---------------- KV^T[v,d] = (1/N) sum_n Kp[n,d] V[n,v]  (WMMA over K=N) ----------------
__global__ __launch_bounds__(256) void attn_kv(const bf16_t* __restrict__ KpT,
                                               const bf16_t* __restrict__ Vt,
                                               bf16_t* __restrict__ KVt)
{
    const int bh = blockIdx.x;              // 0..31
    const int wave = threadIdx.x >> 5;      // 0..7
    const int lane = threadIdx.x & 31;
    const int dt  = wave & 3;               // d tile
    const int vt0 = (wave >> 2) * 2;        // v tiles vt0, vt0+1
    const int row = lane & 15;
    const int kb  = (lane >> 4) * 8;

    const bf16_t* Kb = KpT + (size_t)bh * DH * NN;
    const bf16_t* Vb = Vt  + (size_t)bh * DH * NN;
    const bf16_t* ap  = Kb + (size_t)(dt * 16 + row) * NN + kb;
    const bf16_t* bpA = Vb + (size_t)((vt0 + 0) * 16 + row) * NN + kb;
    const bf16_t* bpB = Vb + (size_t)((vt0 + 1) * 16 + row) * NN + kb;

    v8f acc[2] = {};
    for (int k0 = 0; k0 < NN; k0 += 32) {
        v16bf a  = load_frag(ap  + k0);
        v16bf b0 = load_frag(bpA + k0);
        v16bf b1 = load_frag(bpB + k0);
        acc[0] = __builtin_amdgcn_wmma_f32_16x16x32_bf16(false, a, false, b0, (short)0, acc[0], false, false);
        acc[1] = __builtin_amdgcn_wmma_f32_16x16x32_bf16(false, a, false, b1, (short)0, acc[1], false, false);
    }

    const int crow = (lane >> 4) * 8;
    const int ccol = lane & 15;
    const float inv = 1.0f / (float)NN;
#pragma unroll
    for (int j = 0; j < 2; ++j)
#pragma unroll
        for (int r = 0; r < 8; ++r) {
            int d = dt * 16 + crow + r;
            int v = (vt0 + j) * 16 + ccol;
            KVt[(size_t)bh * (DH * DH) + v * DH + d] = (bf16_t)(acc[j][r] * inv);
        }
}

// ---------------- msg[n,v] = (Q @ KV) * Z * N  (WMMA, K=64) ----------------
__global__ __launch_bounds__(256) void attn_msg(const bf16_t* __restrict__ Qp,
                                                const bf16_t* __restrict__ KVt,
                                                const float* __restrict__ Z,
                                                bf16_t* __restrict__ Msg)
{
    const int bh = blockIdx.y;
    const int b = bh >> 3, h = bh & 7;
    const int wave = threadIdx.x >> 5;
    const int lane = threadIdx.x & 31;
    const int n0 = blockIdx.x * 128 + wave * 16;
    const int row = lane & 15;
    const int kb  = (lane >> 4) * 8;

    const bf16_t* ap  = Qp + ((size_t)(b * NN + n0 + row)) * CC + h * DH + kb;
    const bf16_t* kvb = KVt + (size_t)bh * (DH * DH);

    v8f acc[4] = {};
    for (int k0 = 0; k0 < DH; k0 += 32) {
        v16bf a = load_frag(ap + k0);
#pragma unroll
        for (int j = 0; j < 4; ++j) {
            v16bf bfr = load_frag(kvb + (j * 16 + row) * DH + kb + k0);
            acc[j] = __builtin_amdgcn_wmma_f32_16x16x32_bf16(false, a, false, bfr, (short)0, acc[j], false, false);
        }
    }

    const int crow = (lane >> 4) * 8;
    const int ccol = lane & 15;
#pragma unroll
    for (int j = 0; j < 4; ++j)
#pragma unroll
        for (int r = 0; r < 8; ++r) {
            int n = n0 + crow + r;
            size_t rg = (size_t)b * NN + n;
            float z = Z[rg * HH + h];
            Msg[rg * CC + h * DH + j * 16 + ccol] = (bf16_t)(acc[j][r] * z * (float)NN);
        }
}

// ---------------- LayerNorm 1: merged bf16 -> ffin[:, 512:1024] bf16 ----------------
__global__ __launch_bounds__(256) void ln1_kernel(const bf16_t* __restrict__ In,
                                                  const float* __restrict__ g,
                                                  const float* __restrict__ be,
                                                  bf16_t* __restrict__ ffin)
{
    __shared__ float s1[256], s2[256];
    const int rowI = blockIdx.x;
    const int t = threadIdx.x;
    const bf16_t* p = In + (size_t)rowI * CC;
    float v0 = (float)p[t], v1 = (float)p[t + 256];
    s1[t] = v0 + v1;
    s2[t] = v0 * v0 + v1 * v1;
    __syncthreads();
    for (int ofs = 128; ofs > 0; ofs >>= 1) {
        if (t < ofs) { s1[t] += s1[t + ofs]; s2[t] += s2[t + ofs]; }
        __syncthreads();
    }
    float mu  = s1[0] * (1.0f / CC);
    float var = s2[0] * (1.0f / CC) - mu * mu;
    float rs  = rsqrtf(var + 1e-5f);
    bf16_t* out = ffin + (size_t)rowI * (2 * CC) + CC;
    out[t]       = (bf16_t)((v0 - mu) * rs * g[t]       + be[t]);
    out[t + 256] = (bf16_t)((v1 - mu) * rs * g[t + 256] + be[t + 256]);
}

// ---------------- LayerNorm 2 + residual + transpose to [B,C,N] f32 ----------------
__global__ __launch_bounds__(256) void ln2_kernel(const bf16_t* __restrict__ In,
                                                  const float* __restrict__ g,
                                                  const float* __restrict__ be,
                                                  const float* __restrict__ x,
                                                  float* __restrict__ y)
{
    __shared__ float s1[256], s2[256];
    const int rowI = blockIdx.x;
    const int t = threadIdx.x;
    const bf16_t* p = In + (size_t)rowI * CC;
    float v0 = (float)p[t], v1 = (float)p[t + 256];
    s1[t] = v0 + v1;
    s2[t] = v0 * v0 + v1 * v1;
    __syncthreads();
    for (int ofs = 128; ofs > 0; ofs >>= 1) {
        if (t < ofs) { s1[t] += s1[t + ofs]; s2[t] += s2[t + ofs]; }
        __syncthreads();
    }
    float mu  = s1[0] * (1.0f / CC);
    float var = s2[0] * (1.0f / CC) - mu * mu;
    float rs  = rsqrtf(var + 1e-5f);
    const int b = rowI >> 12, n = rowI & (NN - 1);
    {
        int c = t;
        size_t oi = ((size_t)(b * CC + c)) * NN + n;
        y[oi] = x[oi] + ((v0 - mu) * rs * g[c] + be[c]);
    }
    {
        int c = t + 256;
        size_t oi = ((size_t)(b * CC + c)) * NN + n;
        y[oi] = x[oi] + ((v1 - mu) * rs * g[c] + be[c]);
    }
}

// ---------------- host ----------------
extern "C" void kernel_launch(void* const* d_in, const int* in_sizes, int n_in,
                              void* d_out, int out_size, void* d_ws, size_t ws_size,
                              hipStream_t stream)
{
    const float* x      = (const float*)d_in[0];
    const float* xmask  = (const float*)d_in[1];
    const float* qw     = (const float*)d_in[2];
    const float* kw     = (const float*)d_in[3];
    const float* vw     = (const float*)d_in[4];
    const float* mergew = (const float*)d_in[5];
    const float* n1g    = (const float*)d_in[6];
    const float* n1b    = (const float*)d_in[7];
    const float* ff1w   = (const float*)d_in[8];
    const float* ff2w   = (const float*)d_in[9];
    const float* n2g    = (const float*)d_in[10];
    const float* n2b    = (const float*)d_in[11];
    float* out = (float*)d_out;

    char* ws = (char*)d_ws;
    size_t off = 0;
    auto alloc = [&](size_t bytes) -> void* {
        void* p = ws + off;
        off = (off + bytes + 255) & ~(size_t)255;
        return p;
    };

    bf16_t* ffin = (bf16_t*)alloc((size_t)MM * 1024 * 2);       // [M,1024] concat buffer
    bf16_t* Wqkv = (bf16_t*)alloc((size_t)1536 * 512 * 2);      // packed q/k/v weights
    bf16_t* Wmg  = (bf16_t*)alloc((size_t)512 * 512 * 2);
    bf16_t* Wf1  = (bf16_t*)alloc((size_t)1024 * 1024 * 2);
    bf16_t* Wf2  = (bf16_t*)alloc((size_t)512 * 1024 * 2);
    bf16_t* QKV  = (bf16_t*)alloc((size_t)MM * 1536 * 2);       // reused later
    bf16_t* Qp   = (bf16_t*)alloc((size_t)MM * 512 * 2);        // reused for msg2
    bf16_t* KpT  = (bf16_t*)alloc((size_t)32 * 64 * NN * 2);
    bf16_t* Vt   = (bf16_t*)alloc((size_t)32 * 64 * NN * 2);
    float*  Ksum = (float*) alloc((size_t)32 * 64 * 4);
    float*  Zb   = (float*) alloc((size_t)MM * HH * 4);
    bf16_t* KVt  = (bf16_t*)alloc((size_t)32 * DH * DH * 2);
    bf16_t* msg  = (bf16_t*)alloc((size_t)MM * 512 * 2);
    // region reuse (all launches serialized on one stream):
    bf16_t* merged = QKV;                       // [M,512]  (QKV dead after feature_map)
    bf16_t* hbuf   = QKV + (size_t)MM * 512;    // [M,1024] (disjoint from merged)
    bf16_t* msg2   = Qp;                        // [M,512]  (Qp dead after attn_msg/z)

    const int T = 256;

    // weights -> bf16
    cvt_f32_bf16<<<(262144 + T - 1) / T, T, 0, stream>>>(qw,     Wqkv,           262144);
    cvt_f32_bf16<<<(262144 + T - 1) / T, T, 0, stream>>>(kw,     Wqkv + 262144,  262144);
    cvt_f32_bf16<<<(262144 + T - 1) / T, T, 0, stream>>>(vw,     Wqkv + 524288,  262144);
    cvt_f32_bf16<<<(262144 + T - 1) / T, T, 0, stream>>>(mergew, Wmg,            262144);
    cvt_f32_bf16<<<(1048576 + T - 1) / T, T, 0, stream>>>(ff1w,  Wf1,           1048576);
    cvt_f32_bf16<<<(524288 + T - 1) / T, T, 0, stream>>>(ff2w,   Wf2,            524288);

    // xs = transpose(x) -> ffin[:, 0:512]
    x_to_ffin<<<(MM * CC) / T, T, 0, stream>>>(x, ffin);

    // fused QKV projection: [M,512] x [1536,512]^T
    gemm_bf16_nt<0><<<dim3(MM / 128, 1536 / 64), 256, 0, stream>>>(ffin, 1024, Wqkv, 512, QKV, 1536, 512);

    // feature map + transposed K/V layouts
    feature_map<<<(MM * CC) / T, T, 0, stream>>>(QKV, xmask, Qp, KpT, Vt);
    ksum_kernel<<<(32 * 64) / T, T, 0, stream>>>(KpT, Ksum);
    z_kernel<<<(MM * HH) / T, T, 0, stream>>>(Qp, Ksum, Zb);

    // KV = K^T V / N  and  msg = Q KV Z N   (both WMMA)
    attn_kv<<<32, 256, 0, stream>>>(KpT, Vt, KVt);
    attn_msg<<<dim3(NN / 128, 32), 256, 0, stream>>>(Qp, KVt, Zb, msg);

    // merge projection
    gemm_bf16_nt<0><<<dim3(MM / 128, 512 / 64), 256, 0, stream>>>(msg, 512, Wmg, 512, merged, 512, 512);
    ln1_kernel<<<MM, 256, 0, stream>>>(merged, n1g, n1b, ffin);

    // FFN
    gemm_bf16_nt<1><<<dim3(MM / 128, 1024 / 64), 256, 0, stream>>>(ffin, 1024, Wf1, 1024, hbuf, 1024, 1024);
    gemm_bf16_nt<0><<<dim3(MM / 128, 512 / 64), 256, 0, stream>>>(hbuf, 1024, Wf2, 1024, msg2, 512, 1024);

    // LN2 + residual + transpose out
    ln2_kernel<<<MM, 256, 0, stream>>>(msg2, n2g, n2b, x, out);
}
